// AVPViT_1082331758904
// MI455X (gfx1250) — matrix operand order repaired and checked by reading, not compiled, not run on hardware
//
#include <hip/hip_runtime.h>
#include <math.h>

#define B_   128
#define D_   768
#define NH_  12
#define HD_  64
#define NL_  54
#define NS_  256
#define L_   12
#define SCALE_ 0.125f   /* 1/sqrt(64) */

typedef __bf16 bf16_t;
typedef bf16_t v16bf __attribute__((ext_vector_type(16)));
typedef float  v8f   __attribute__((ext_vector_type(8)));

static __device__ __forceinline__ bf16_t f2bf(float f) {
  unsigned u = __builtin_bit_cast(unsigned, f);
  u += 0x7FFFu + ((u >> 16) & 1u);          // round-to-nearest-even
  unsigned short h = (unsigned short)(u >> 16);
  return __builtin_bit_cast(bf16_t, h);
}

static __device__ __forceinline__ float gelu_f(float x) {
  float x3 = x * x * x;
  return 0.5f * x * (1.f + tanhf(0.7978845608028654f * (x + 0.044715f * x3)));
}

// ---------------------------------------------------------------------------
// Batched strided WMMA GEMM:  C = epi(alpha * A[M,K] @ B)
//   BT=0: B stored row-major [K,N];  BT=1: B stored [N,K] (i.e. C = A @ B^T)
//   blockIdx.z = batch*heads + head; per-operand batch/head strides.
// Tile: 128x128 per block (8 waves, each 32x64 via 2x4 of 16x16x32 bf16 WMMA)
// ---------------------------------------------------------------------------
#define BM   128
#define BN   128
#define BKT  64
#define LDSK 72   // padded K stride in LDS (bf16 elements), 144B rows

template<int EPI, bool BT>
__global__ void __launch_bounds__(256) gemm_wmma_k(
    const float* __restrict__ A, const float* __restrict__ Bm, float* __restrict__ C,
    int M, int N, int K,
    long sAb, long sAh, long sBb, long sBh, long sCb, long sCh, int heads,
    int lda, int ldb, int ldc, float alpha, const float* __restrict__ gate)
{
  __shared__ bf16_t As[BM][LDSK];   // [m][k]
  __shared__ bf16_t Bs[BN][LDSK];   // [n][k]

  const int z  = blockIdx.z;
  const int zb = z / heads;
  const int zh = z - zb * heads;
  A  += (long)zb * sAb + (long)zh * sAh;
  Bm += (long)zb * sBb + (long)zh * sBh;
  C  += (long)zb * sCb + (long)zh * sCh;

  const int m0   = blockIdx.y * BM;
  const int n0   = blockIdx.x * BN;
  const int tid  = threadIdx.x;
  const int lane = tid & 31;
  const int wid  = tid >> 5;
  const int wm   = wid >> 1;      // 0..3 (M direction)
  const int wn   = wid & 1;       // 0..1 (N direction)
  const int lh   = lane >> 4;     // half-wave select
  const int ll   = lane & 15;

  v8f acc[2][4];
#pragma unroll
  for (int i = 0; i < 2; ++i)
#pragma unroll
    for (int j = 0; j < 4; ++j)
#pragma unroll
      for (int e = 0; e < 8; ++e) acc[i][j][e] = 0.f;

  const bool fa = ((lda & 3) == 0);
  const bool fb = ((ldb & 3) == 0);

  for (int k0 = 0; k0 < K; k0 += BKT) {
    // ---- stage A tile (128x64 f32 -> bf16) ----
#pragma unroll
    for (int it = 0; it < 8; ++it) {
      int vid = tid + it * 256;
      int r   = vid >> 4;
      int c4  = (vid & 15) << 2;
      int gm  = m0 + r, gk = k0 + c4;
      float x0 = 0.f, x1 = 0.f, x2 = 0.f, x3 = 0.f;
      if (gm < M) {
        const float* p = A + (long)gm * lda + gk;
        if (fa && gk + 3 < K) { float4 f = *(const float4*)p; x0 = f.x; x1 = f.y; x2 = f.z; x3 = f.w; }
        else {
          if (gk     < K) x0 = p[0];
          if (gk + 1 < K) x1 = p[1];
          if (gk + 2 < K) x2 = p[2];
          if (gk + 3 < K) x3 = p[3];
        }
      }
      As[r][c4] = f2bf(x0); As[r][c4 + 1] = f2bf(x1); As[r][c4 + 2] = f2bf(x2); As[r][c4 + 3] = f2bf(x3);
    }
    // ---- stage B tile into Bs[n][k] ----
    if (BT) {
#pragma unroll
      for (int it = 0; it < 8; ++it) {
        int vid = tid + it * 256;
        int r   = vid >> 4;            // n
        int c4  = (vid & 15) << 2;     // k
        int gn  = n0 + r, gk = k0 + c4;
        float x0 = 0.f, x1 = 0.f, x2 = 0.f, x3 = 0.f;
        if (gn < N) {
          const float* p = Bm + (long)gn * ldb + gk;
          if (fb && gk + 3 < K) { float4 f = *(const float4*)p; x0 = f.x; x1 = f.y; x2 = f.z; x3 = f.w; }
          else {
            if (gk     < K) x0 = p[0];
            if (gk + 1 < K) x1 = p[1];
            if (gk + 2 < K) x2 = p[2];
            if (gk + 3 < K) x3 = p[3];
          }
        }
        Bs[r][c4] = f2bf(x0); Bs[r][c4 + 1] = f2bf(x1); Bs[r][c4 + 2] = f2bf(x2); Bs[r][c4 + 3] = f2bf(x3);
      }
    } else {
#pragma unroll
      for (int it = 0; it < 8; ++it) {
        int vid = tid + it * 256;
        int kr  = vid >> 5;            // k (0..63)
        int c4  = (vid & 31) << 2;     // n (0..124)
        int gk  = k0 + kr, gn = n0 + c4;
        float x0 = 0.f, x1 = 0.f, x2 = 0.f, x3 = 0.f;
        if (gk < K) {
          const float* p = Bm + (long)gk * ldb + gn;
          if (fb && gn + 3 < N) { float4 f = *(const float4*)p; x0 = f.x; x1 = f.y; x2 = f.z; x3 = f.w; }
          else {
            if (gn     < N) x0 = p[0];
            if (gn + 1 < N) x1 = p[1];
            if (gn + 2 < N) x2 = p[2];
            if (gn + 3 < N) x3 = p[3];
          }
        }
        Bs[c4][kr] = f2bf(x0); Bs[c4 + 1][kr] = f2bf(x1); Bs[c4 + 2][kr] = f2bf(x2); Bs[c4 + 3][kr] = f2bf(x3);
      }
    }
    __syncthreads();

    // prefetch next K tile into cache while compute runs (global_prefetch_b8)
    if (k0 + BKT < K) {
      if (tid < BM && m0 + tid < M) __builtin_prefetch(A + (long)(m0 + tid) * lda + k0 + BKT, 0, 1);
      if (BT) { if (tid < BN && n0 + tid < N) __builtin_prefetch(Bm + (long)(n0 + tid) * ldb + k0 + BKT, 0, 1); }
      else    { if (tid < BKT && k0 + BKT + tid < K) __builtin_prefetch(Bm + (long)(k0 + BKT + tid) * ldb + n0, 0, 1); }
    }

#pragma unroll
    for (int kk = 0; kk < BKT; kk += 32) {
      // A fragment: lanes 0-15 hold K {kk..kk+7, kk+16..kk+23}; lanes 16-31 the other halves
      v16bf af[2];
#pragma unroll
      for (int mt = 0; mt < 2; ++mt) {
        const bf16_t* p = &As[wm * 32 + mt * 16 + ll][kk + lh * 8];
#pragma unroll
        for (int e = 0; e < 8; ++e) { af[mt][e] = p[e]; af[mt][8 + e] = p[16 + e]; }
      }
      // B fragment: lanes 0-15 hold K {kk..kk+7, kk+16..kk+23}; lanes 16-31 the other halves, for column ll
      v16bf bq[4];
#pragma unroll
      for (int nt = 0; nt < 4; ++nt) {
        const bf16_t* p = &Bs[wn * 64 + nt * 16 + ll][kk + lh * 8];
#pragma unroll
        for (int e = 0; e < 8; ++e) { bq[nt][e] = p[e]; bq[nt][8 + e] = p[16 + e]; }
      }
#pragma unroll
      for (int mt = 0; mt < 2; ++mt)
#pragma unroll
        for (int nt = 0; nt < 4; ++nt)
          acc[mt][nt] = __builtin_amdgcn_wmma_f32_16x16x32_bf16(
              false, af[mt], false, bq[nt], (short)0, acc[mt][nt], false, false);
    }
    __syncthreads();
  }

  // epilogue: VGPR r -> M = r + 8*lh, N = ll (per ISA C/D layout)
#pragma unroll
  for (int mt = 0; mt < 2; ++mt) {
#pragma unroll
    for (int nt = 0; nt < 4; ++nt) {
      int col = n0 + wn * 64 + nt * 16 + ll;
#pragma unroll
      for (int r = 0; r < 8; ++r) {
        int gm = m0 + wm * 32 + mt * 16 + lh * 8 + r;
        if (gm < M && col < N) {
          float v = acc[mt][nt][r];
          long  o = (long)gm * ldc + col;
          if      (EPI == 0) C[o]  = alpha * v;
          else if (EPI == 1) C[o]  = gelu_f(v);
          else if (EPI == 2) C[o] += v;
          else               C[o] += gate[col] * v;
        }
      }
    }
  }
}

// ---------------------------------------------------------------------------
// Elementwise kernels
// ---------------------------------------------------------------------------
__global__ void bcast_k(const float* __restrict__ s, float* __restrict__ d, long n, long mod) {
  long i = (long)blockIdx.x * 256 + threadIdx.x;
  if (i < n) d[i] = s[i % mod];
}

__global__ void rope_table_k(const float* __restrict__ centers, const float* __restrict__ scales,
                             float* __restrict__ so, float* __restrict__ co,
                             int h, int w, int perBatch) {
  int nt = h * w;
  long total = (long)(perBatch ? B_ : 1) * nt * 32;
  long gid = (long)blockIdx.x * 256 + threadIdx.x;
  if (gid >= total) return;
  int j = (int)(gid & 31);
  long r = gid >> 5;
  int t = (int)(r % nt);
  int b = (int)(r / nt);
  int c = j >> 4, f = j & 15;
  float gv = (c == 0) ? (((float)(t / w) + 0.5f) / (float)h * 2.f - 1.f)
                      : (((float)(t % w) + 0.5f) / (float)w * 2.f - 1.f);
  float pos = perBatch ? (centers[b * 2 + c] + scales[b] * gv) : gv;
  float ang = pos * powf(100.f, -(float)f / 16.f);
  so[gid] = sinf(ang);
  co[gid] = cosf(ang);
}

__global__ void rope_apply_k(float* __restrict__ x, const float* __restrict__ sn,
                             const float* __restrict__ cs, int T, int prefix,
                             int ld, int off, int perBatch) {
  int Tr = T - prefix;
  long total = (long)B_ * Tr * NH_ * 32;
  long gid = (long)blockIdx.x * 256 + threadIdx.x;
  if (gid >= total) return;
  int j = (int)(gid & 31);
  long r1 = gid >> 5;
  int h = (int)(r1 % NH_);
  long r2 = r1 / NH_;
  int t = (int)(r2 % Tr);
  int b = (int)(r2 / Tr);
  long base = ((long)b * T + prefix + t) * ld + off + h * HD_;
  long si = ((perBatch ? (long)b * Tr : 0) + t) * 32 + j;
  float s = sn[si], c = cs[si];
  float x1 = x[base + j], x2 = x[base + 32 + j];
  x[base + j]      = x1 * c - x2 * s;
  x[base + 32 + j] = x2 * c + x1 * s;
}

__global__ void __launch_bounds__(256) layernorm_k(const float* __restrict__ x,
    const float* __restrict__ g, const float* __restrict__ bb, float* __restrict__ o) {
  __shared__ float red[256];
  int tid = threadIdx.x;
  long row = blockIdx.x;
  const float* xr = x + row * D_;
  float a0 = xr[tid], a1 = xr[tid + 256], a2 = xr[tid + 512];
  red[tid] = a0 + a1 + a2;
  __syncthreads();
  for (int s = 128; s > 0; s >>= 1) { if (tid < s) red[tid] += red[tid + s]; __syncthreads(); }
  float m = red[0] * (1.f / (float)D_);
  __syncthreads();
  float d0 = a0 - m, d1 = a1 - m, d2 = a2 - m;
  red[tid] = d0 * d0 + d1 * d1 + d2 * d2;
  __syncthreads();
  for (int s = 128; s > 0; s >>= 1) { if (tid < s) red[tid] += red[tid + s]; __syncthreads(); }
  float rs = rsqrtf(red[0] * (1.f / (float)D_) + 1e-6f);
  float* orow = o + row * D_;
  orow[tid]       = d0 * rs * g[tid]       + bb[tid];
  orow[tid + 256] = d1 * rs * g[tid + 256] + bb[tid + 256];
  orow[tid + 512] = d2 * rs * g[tid + 512] + bb[tid + 512];
}

__global__ void __launch_bounds__(256) softmax_k(float* __restrict__ s, int Tk) {
  __shared__ float red[256];
  int tid = threadIdx.x;
  float* p = s + (long)blockIdx.x * Tk;
  float v = (tid < Tk) ? p[tid] : -3.4e38f;
  red[tid] = v;
  __syncthreads();
  for (int st = 128; st > 0; st >>= 1) { if (tid < st) red[tid] = fmaxf(red[tid], red[tid + st]); __syncthreads(); }
  float mx = red[0];
  __syncthreads();
  float e = (tid < Tk) ? __expf(v - mx) : 0.f;
  red[tid] = e;
  __syncthreads();
  for (int st = 128; st > 0; st >>= 1) { if (tid < st) red[tid] += red[tid + st]; __syncthreads(); }
  float inv = 1.f / red[0];
  if (tid < Tk) p[tid] = e * inv;
}

// ---------------------------------------------------------------------------
// Host-side launcher helpers
// ---------------------------------------------------------------------------
static inline void gemm(hipStream_t st, const float* A, const float* Bm, float* C,
                        int M, int N, int K,
                        long sAb, long sAh, long sBb, long sBh, long sCb, long sCh,
                        int heads, int batches,
                        int lda, int ldb, int ldc, float alpha, const float* gate,
                        int epi, bool bt) {
  dim3 g((unsigned)((N + BN - 1) / BN), (unsigned)((M + BM - 1) / BM), (unsigned)batches);
  dim3 b(256, 1, 1);
#define GL(E, T) gemm_wmma_k<E, T><<<g, b, 0, st>>>(A, Bm, C, M, N, K, sAb, sAh, sBb, sBh, sCb, sCh, heads, lda, ldb, ldc, alpha, gate)
  if (!bt) {
    switch (epi) { case 0: GL(0, false); break; case 1: GL(1, false); break; case 2: GL(2, false); break; default: GL(3, false); }
  } else {
    switch (epi) { case 0: GL(0, true); break; case 1: GL(1, true); break; case 2: GL(2, true); break; default: GL(3, true); }
  }
#undef GL
}

static inline void proj(hipStream_t st, const float* A, const float* W, float* C,
                        int M, int N, int K, float alpha, const float* gate, int epi) {
  gemm(st, A, W, C, M, N, K, 0, 0, 0, 0, 0, 0, 1, 1, K, N, N, alpha, gate, epi, false);
}

extern "C" void kernel_launch(void* const* d_in, const int* in_sizes, int n_in,
                              void* d_out, int out_size, void* d_ws, size_t ws_size,
                              hipStream_t stream) {
  (void)in_sizes; (void)n_in; (void)ws_size;
  const float* in_local   = (const float*)d_in[0];
  const float* centers    = (const float*)d_in[1];
  const float* scales     = (const float*)d_in[2];
  const float* scene_tok  = (const float*)d_in[3];
  const float* read_gate  = (const float*)d_in[4];
  const float* write_gate = (const float*)d_in[5];
  const float* read_Wq    = (const float*)d_in[6];
  const float* read_Wk    = (const float*)d_in[7];
  const float* read_Wv    = (const float*)d_in[8];
  const float* read_Wo    = (const float*)d_in[9];
  const float* write_Wq   = (const float*)d_in[10];
  const float* write_Wk   = (const float*)d_in[11];
  const float* write_Wv   = (const float*)d_in[12];
  const float* write_Wo   = (const float*)d_in[13];
  const float* ln1_g      = (const float*)d_in[14];
  const float* ln1_b      = (const float*)d_in[15];
  const float* qkv_W      = (const float*)d_in[16];
  const float* attn_Wo    = (const float*)d_in[17];
  const float* ln2_g      = (const float*)d_in[18];
  const float* ln2_b      = (const float*)d_in[19];
  const float* mlp_W1     = (const float*)d_in[20];
  const float* mlp_W2     = (const float*)d_in[21];

  const size_t LND = (size_t)B_ * NL_ * D_;   // 5,308,416
  const size_t SND = (size_t)B_ * NS_ * D_;   // 25,165,824
  const size_t DD  = (size_t)D_ * D_;

  float* w = (float*)d_ws;
  size_t off = 0;
  float* loc  = w + off; off += LND;
  float* scn  = w + off; off += SND;
  float* lsin = w + off; off += (size_t)B_ * 49 * 32;
  float* lcos = w + off; off += (size_t)B_ * 49 * 32;
  float* ssin = w + off; off += (size_t)NS_ * 32;
  float* scos = w + off; off += (size_t)NS_ * 32;
  float* hb   = w + off; off += LND;                       // LN outputs
  float* tq   = w + off; off += SND;                       // q / qkv buffer
  float* tk   = w + off; off += SND;                       // k buffer
  float* tv   = w + off; off += SND;                       // v buffer
  float* sc   = w + off; off += (size_t)B_ * NL_ * 4 * D_; // scores / mlp hidden (21.2M floats)
  float* ao   = w + off; off += SND;                       // merged attention output

  hipMemcpyAsync(loc, in_local, LND * sizeof(float), hipMemcpyDeviceToDevice, stream);
  {
    long tot = (long)SND;
    bcast_k<<<(unsigned)((tot + 255) / 256), 256, 0, stream>>>(scene_tok, scn, tot, (long)NS_ * D_);
  }
  rope_table_k<<<(B_ * 49 * 32 + 255) / 256, 256, 0, stream>>>(centers, scales, lsin, lcos, 7, 7, 1);
  rope_table_k<<<(NS_ * 32 + 255) / 256, 256, 0, stream>>>(nullptr, nullptr, ssin, scos, 16, 16, 0);

  const int Mloc = B_ * NL_;   // 6912 (multiple of 128)
  const int Mscn = B_ * NS_;   // 32768
  const int BH   = B_ * NH_;   // 1536

  for (int i = 0; i < L_; ++i) {
    // ---------------- READ cross-attention: local <- scene ----------------
    proj(stream, loc, read_Wq + (size_t)i * DD, tq, Mloc, D_, D_, 1.f, nullptr, 0);
    proj(stream, scn, read_Wk + (size_t)i * DD, tk, Mscn, D_, D_, 1.f, nullptr, 0);
    proj(stream, scn, read_Wv + (size_t)i * DD, tv, Mscn, D_, D_, 1.f, nullptr, 0);
    rope_apply_k<<<(B_ * 49 * NH_ * 32 + 255) / 256, 256, 0, stream>>>(tq, lsin, lcos, NL_, 5, D_, 0, 1);
    rope_apply_k<<<(B_ * NS_ * NH_ * 32 + 255) / 256, 256, 0, stream>>>(tk, ssin, scos, NS_, 0, D_, 0, 0);
    gemm(stream, tq, tk, sc, NL_, NS_, HD_,
         (long)NL_ * D_, HD_, (long)NS_ * D_, HD_, (long)NH_ * NL_ * NS_, (long)NL_ * NS_,
         NH_, BH, D_, D_, NS_, SCALE_, nullptr, 0, true);
    softmax_k<<<BH * NL_, 256, 0, stream>>>(sc, NS_);
    gemm(stream, sc, tv, ao, NL_, HD_, NS_,
         (long)NH_ * NL_ * NS_, (long)NL_ * NS_, (long)NS_ * D_, HD_, (long)NL_ * D_, HD_,
         NH_, BH, NS_, D_, D_, 1.f, nullptr, 0, false);
    proj(stream, ao, read_Wo + (size_t)i * DD, loc, Mloc, D_, D_, 1.f, read_gate + (size_t)i * D_, 3);

    // ---------------- ViT block on local ----------------
    layernorm_k<<<Mloc, 256, 0, stream>>>(loc, ln1_g + (size_t)i * D_, ln1_b + (size_t)i * D_, hb);
    proj(stream, hb, qkv_W + (size_t)i * DD * 3, tq, Mloc, 3 * D_, D_, 1.f, nullptr, 0);
    rope_apply_k<<<(B_ * 49 * NH_ * 32 + 255) / 256, 256, 0, stream>>>(tq, lsin, lcos, NL_, 5, 3 * D_, 0, 1);
    rope_apply_k<<<(B_ * 49 * NH_ * 32 + 255) / 256, 256, 0, stream>>>(tq, lsin, lcos, NL_, 5, 3 * D_, D_, 1);
    gemm(stream, tq, tq + D_, sc, NL_, NL_, HD_,
         (long)NL_ * 3 * D_, HD_, (long)NL_ * 3 * D_, HD_, (long)NH_ * NL_ * NL_, (long)NL_ * NL_,
         NH_, BH, 3 * D_, 3 * D_, NL_, SCALE_, nullptr, 0, true);
    softmax_k<<<BH * NL_, 256, 0, stream>>>(sc, NL_);
    gemm(stream, sc, tq + 2 * D_, ao, NL_, HD_, NL_,
         (long)NH_ * NL_ * NL_, (long)NL_ * NL_, (long)NL_ * 3 * D_, HD_, (long)NL_ * D_, HD_,
         NH_, BH, NL_, 3 * D_, D_, 1.f, nullptr, 0, false);
    proj(stream, ao, attn_Wo + (size_t)i * DD, loc, Mloc, D_, D_, 1.f, nullptr, 2);
    layernorm_k<<<Mloc, 256, 0, stream>>>(loc, ln2_g + (size_t)i * D_, ln2_b + (size_t)i * D_, hb);
    proj(stream, hb, mlp_W1 + (size_t)i * DD * 4, sc, Mloc, 4 * D_, D_, 1.f, nullptr, 1);
    proj(stream, sc, mlp_W2 + (size_t)i * DD * 4, loc, Mloc, D_, 4 * D_, 1.f, nullptr, 2);

    // ---------------- WRITE cross-attention: scene <- local ----------------
    proj(stream, scn, write_Wq + (size_t)i * DD, tq, Mscn, D_, D_, 1.f, nullptr, 0);
    proj(stream, loc, write_Wk + (size_t)i * DD, tk, Mloc, D_, D_, 1.f, nullptr, 0);
    proj(stream, loc, write_Wv + (size_t)i * DD, tv, Mloc, D_, D_, 1.f, nullptr, 0);
    rope_apply_k<<<(B_ * NS_ * NH_ * 32 + 255) / 256, 256, 0, stream>>>(tq, ssin, scos, NS_, 0, D_, 0, 0);
    rope_apply_k<<<(B_ * 49 * NH_ * 32 + 255) / 256, 256, 0, stream>>>(tk, lsin, lcos, NL_, 5, D_, 0, 1);
    gemm(stream, tq, tk, sc, NS_, NL_, HD_,
         (long)NS_ * D_, HD_, (long)NL_ * D_, HD_, (long)NH_ * NS_ * NL_, (long)NS_ * NL_,
         NH_, BH, D_, D_, NL_, SCALE_, nullptr, 0, true);
    softmax_k<<<BH * NS_, 256, 0, stream>>>(sc, NL_);
    gemm(stream, sc, tv, ao, NS_, HD_, NL_,
         (long)NH_ * NS_ * NL_, (long)NS_ * NL_, (long)NL_ * D_, HD_, (long)NS_ * D_, HD_,
         NH_, BH, NL_, D_, D_, 1.f, nullptr, 0, false);
    proj(stream, ao, write_Wo + (size_t)i * DD, scn, Mscn, D_, D_, 1.f, write_gate + (size_t)i * D_, 3);
  }

  hipMemcpyAsync(d_out, loc, LND * sizeof(float), hipMemcpyDeviceToDevice, stream);
  hipMemcpyAsync((float*)d_out + LND, scn, SND * sizeof(float), hipMemcpyDeviceToDevice, stream);
}
